// GQA_14688788152667
// MI455X (gfx1250) — compile-verified, matrix operand order checked by reading, not run on hardware
//
#include <hip/hip_runtime.h>
#include <stdint.h>

// Problem constants (from reference): B=2, S=2048, E=2048, H=16, KVH=4, D=128, G=4
#define BB   2
#define SS   2048
#define EE   2048
#define HH   16
#define KVH  4
#define DD   128
#define FQKV 3072          // H*D + 2*KVH*D
#define BS   (BB * SS)     // 4096 rows

typedef __attribute__((ext_vector_type(16))) __bf16 v16bf;
typedef __attribute__((ext_vector_type(8)))  float  v8f;

union FragU { v16bf v; uint4 u[2]; };

__device__ __forceinline__ uint16_t f2bf(float f) {
  uint32_t x = __float_as_uint(f);
  return (uint16_t)((x + 0x7FFFu + ((x >> 16) & 1u)) >> 16);   // round-nearest-even
}
__device__ __forceinline__ float bf2f(uint16_t b) {
  return __uint_as_float(((uint32_t)b) << 16);
}

// WMMA 16x16x32 bf16 operand fragment (wave32).
// A-layout: lane&15 = matrix row, hi = lane>>4; elements = two contiguous
// 8-element K chunks at k0+8*hi and k0+16+8*hi (ISA 7.12.2 16-bit A table).
// B uses the same layout with lane&15 = output column N, reading rows of the
// K-major (i.e. transposed) matrix.
__device__ __forceinline__ v16bf load_frag(const uint16_t* base, int row, int k0,
                                           int ld, int hi) {
  const uint16_t* p = base + (size_t)row * (size_t)ld + (size_t)(k0 + 8 * hi);
  FragU f;
  f.u[0] = *(const uint4*)(p);
  f.u[1] = *(const uint4*)(p + 16);
  return f.v;
}

__device__ __forceinline__ v8f wmma_bf16(v16bf a, v16bf b, v8f c) {
  return __builtin_amdgcn_wmma_f32_16x16x32_bf16(false, a, false, b, (short)0, c,
                                                 false, false);
}

__device__ __forceinline__ v8f zero8() {
  v8f z;
#pragma unroll
  for (int r = 0; r < 8; ++r) z[r] = 0.0f;
  return z;
}

__device__ __forceinline__ float red_max16(float v) {
#pragma unroll
  for (int m = 1; m <= 8; m <<= 1) v = fmaxf(v, __shfl_xor(v, m, 32));
  return v;
}
__device__ __forceinline__ float red_sum16(float v) {
#pragma unroll
  for (int m = 1; m <= 8; m <<= 1) v += __shfl_xor(v, m, 32);
  return v;
}

// ---------------------------------------------------------------- f32 -> bf16
__global__ void cvt_bf16_kernel(const float* __restrict__ in,
                                uint16_t* __restrict__ out, int n) {
  int i = blockIdx.x * blockDim.x + threadIdx.x;
  if (i < n) out[i] = f2bf(in[i]);
}

// ------------------------------------------------- QKV GEMM: [4096,3072] tile
// C = X(bf16)[4096,2048] * Wqkv^T, W stored [3072,2048] row-major (K-major).
// Epilogue scatters into q[B,H,S,D], k[B,KVH,S,D], vT[B,KVH,D,S] (bf16).
__global__ __launch_bounds__(128) void qkv_gemm_kernel(
    const uint16_t* __restrict__ xb, const uint16_t* __restrict__ wb,
    uint16_t* __restrict__ qb, uint16_t* __restrict__ kb,
    uint16_t* __restrict__ vt) {
  const int lane = threadIdx.x & 31, wave = threadIdx.x >> 5;
  const int tile = blockIdx.x * 4 + wave;            // 64 x 48 = 3072 tiles
  const int tm = tile / 48, tn = tile % 48;
  const int m0 = tm * 64, n0 = tn * 64;
  const int l15 = lane & 15, hi = lane >> 4;

  v8f acc[4][4];
#pragma unroll
  for (int i = 0; i < 4; ++i)
#pragma unroll
    for (int j = 0; j < 4; ++j) acc[i][j] = zero8();

  for (int k0 = 0; k0 < EE; k0 += 32) {
    __builtin_prefetch(xb + (size_t)(m0 + l15) * EE + k0 + 256, 0, 0);
    __builtin_prefetch(wb + (size_t)(n0 + l15) * EE + k0 + 256, 0, 0);
    v16bf af[4], bfr[4];
#pragma unroll
    for (int i = 0; i < 4; ++i) af[i] = load_frag(xb, m0 + 16 * i + l15, k0, EE, hi);
#pragma unroll
    for (int j = 0; j < 4; ++j) bfr[j] = load_frag(wb, n0 + 16 * j + l15, k0, EE, hi);
#pragma unroll
    for (int i = 0; i < 4; ++i)
#pragma unroll
      for (int j = 0; j < 4; ++j) acc[i][j] = wmma_bf16(af[i], bfr[j], acc[i][j]);
  }

#pragma unroll
  for (int i = 0; i < 4; ++i)
#pragma unroll
    for (int j = 0; j < 4; ++j) {
      const int col = n0 + 16 * j + l15;             // f index in [0,3072)
      const int kvh = col / 768;
      const int rem = col - kvh * 768;
      const int jj = rem >> 7;                       // 0..5 (0..3=q, 4=k, 5=v)
      const int d = rem & 127;
#pragma unroll
      for (int r = 0; r < 8; ++r) {
        const int row = m0 + 16 * i + r + 8 * hi;    // b*S + s
        const int b = row >> 11, s = row & (SS - 1);
        const uint16_t bv = f2bf(acc[i][j][r]);
        if (jj < 4) {
          const int h = kvh * 4 + jj;
          qb[(((size_t)(b * HH + h)) * SS + s) * DD + d] = bv;
        } else if (jj == 4) {
          kb[(((size_t)(b * KVH + kvh)) * SS + s) * DD + d] = bv;
        } else {
          vt[(((size_t)(b * KVH + kvh)) * DD + d) * SS + s] = bv;
        }
      }
    }
}

// ----------------------------------------------------- interleaved RoPE (q,k)
__global__ void rope_kernel(uint16_t* __restrict__ qb, uint16_t* __restrict__ kb) {
  const int tid = blockIdx.x * blockDim.x + threadIdx.x;
  const int nqrow = BB * HH * SS;                    // 65536
  const int nkrow = BB * KVH * SS;                   // 16384
  if (tid >= (nqrow + nkrow) * 64) return;
  const int r = tid >> 6, i = tid & 63;
  uint16_t* base;
  int s;
  if (r < nqrow) { base = qb + (size_t)r * DD; s = r & (SS - 1); }
  else { base = kb + (size_t)(r - nqrow) * DD; s = (r - nqrow) & (SS - 1); }
  const float inv = __expf(-(float)i * (9.210340371976184f / 64.0f)); // base^(-2i/D)
  const float ang = (float)s * inv;
  float sn, cs;
  sincosf(ang, &sn, &cs);
  const float x1 = bf2f(base[2 * i]), x2 = bf2f(base[2 * i + 1]);
  base[2 * i]     = f2bf(x1 * cs - x2 * sn);
  base[2 * i + 1] = f2bf(x1 * sn + x2 * cs);
}

// --------------------------------------------- flash attention, 32 q rows/wave
__global__ __launch_bounds__(128) void attn_kernel(
    const uint16_t* __restrict__ qb, const uint16_t* __restrict__ kb,
    const uint16_t* __restrict__ vt, uint16_t* __restrict__ ctxb) {
  __shared__ uint16_t pSh[4][32 * 40];
  const int lane = threadIdx.x & 31, wave = threadIdx.x >> 5;
  const int gw = blockIdx.x * 4 + wave;              // 2048 waves total
  const int qt = gw & 63, bh = gw >> 6;
  const int h = bh & (HH - 1), b = bh >> 4;
  const int kvh = h >> 2;
  const int l15 = lane & 15, hi = lane >> 4;
  const int q0 = qt * 32;
  const uint16_t* Q = qb + (size_t)(b * HH + h) * SS * DD;
  const uint16_t* K = kb + (size_t)(b * KVH + kvh) * SS * DD;
  const uint16_t* V = vt + (size_t)(b * KVH + kvh) * DD * SS;  // [D][S]
  uint16_t* P = &pSh[wave][0];
  const float scale = 0.08838834764831845f;          // 1/sqrt(128)

  v16bf qf[2][4];
#pragma unroll
  for (int mi = 0; mi < 2; ++mi)
#pragma unroll
    for (int ks = 0; ks < 4; ++ks)
      qf[mi][ks] = load_frag(Q, q0 + 16 * mi + l15, ks * 32, DD, hi);

  v8f O[2][8];
  float mrow[2][8], lrow[2][8];
#pragma unroll
  for (int mi = 0; mi < 2; ++mi) {
#pragma unroll
    for (int df = 0; df < 8; ++df) O[mi][df] = zero8();
#pragma unroll
    for (int r = 0; r < 8; ++r) { mrow[mi][r] = -3.0e38f; lrow[mi][r] = 0.0f; }
  }

  for (int t = 0; t <= qt; ++t) {
    const int kv0 = t * 32;
    v8f sf[2][2];
#pragma unroll
    for (int mi = 0; mi < 2; ++mi)
#pragma unroll
      for (int nj = 0; nj < 2; ++nj) sf[mi][nj] = zero8();
#pragma unroll
    for (int ks = 0; ks < 4; ++ks) {
      v16bf b0 = load_frag(K, kv0 + l15, ks * 32, DD, hi);
      v16bf b1 = load_frag(K, kv0 + 16 + l15, ks * 32, DD, hi);
#pragma unroll
      for (int mi = 0; mi < 2; ++mi) {
        sf[mi][0] = wmma_bf16(qf[mi][ks], b0, sf[mi][0]);
        sf[mi][1] = wmma_bf16(qf[mi][ks], b1, sf[mi][1]);
      }
    }
    const bool diag = (t == qt);
#pragma unroll
    for (int mi = 0; mi < 2; ++mi)
#pragma unroll
      for (int r = 0; r < 8; ++r) {
        float s0 = sf[mi][0][r] * scale;
        float s1 = sf[mi][1][r] * scale;
        if (diag) {
          const int rowl = 16 * mi + r + 8 * hi;     // local q row
          if (l15 > rowl) s0 = -1.0e30f;
          if (16 + l15 > rowl) s1 = -1.0e30f;
        }
        const float mx = red_max16(fmaxf(s0, s1));
        const float mnew = fmaxf(mrow[mi][r], mx);
        const float alpha = __expf(mrow[mi][r] - mnew);
        const float p0 = __expf(s0 - mnew);
        const float p1 = __expf(s1 - mnew);
        const float sum = red_sum16(p0 + p1);
        lrow[mi][r] = lrow[mi][r] * alpha + sum;
        mrow[mi][r] = mnew;
#pragma unroll
        for (int df = 0; df < 8; ++df) O[mi][df][r] *= alpha;
        const int prow = (16 * mi + r + 8 * hi) * 40;
        P[prow + l15] = f2bf(p0);
        P[prow + 16 + l15] = f2bf(p1);
      }
    asm volatile("s_wait_dscnt 0" ::: "memory");     // P stores visible to reloads
    v16bf pa[2];
#pragma unroll
    for (int mi = 0; mi < 2; ++mi) pa[mi] = load_frag(P, 16 * mi + l15, 0, 40, hi);
#pragma unroll
    for (int df = 0; df < 8; ++df) {
      v16bf vb = load_frag(V, df * 16 + l15, kv0, SS, hi);
#pragma unroll
      for (int mi = 0; mi < 2; ++mi) O[mi][df] = wmma_bf16(pa[mi], vb, O[mi][df]);
    }
  }

#pragma unroll
  for (int mi = 0; mi < 2; ++mi)
#pragma unroll
    for (int df = 0; df < 8; ++df)
#pragma unroll
      for (int r = 0; r < 8; ++r) {
        const int s = q0 + 16 * mi + r + 8 * hi;
        const float o = O[mi][df][r] / lrow[mi][r];
        ctxb[((size_t)(b * SS + s)) * (HH * DD) + h * DD + df * 16 + l15] = f2bf(o);
      }
}

// ------------------------------------------ output GEMM: out = ctx * wo^T (f32)
__global__ __launch_bounds__(128) void out_gemm_kernel(
    const uint16_t* __restrict__ ctxb, const uint16_t* __restrict__ wob,
    float* __restrict__ out) {
  const int lane = threadIdx.x & 31, wave = threadIdx.x >> 5;
  const int tile = blockIdx.x * 4 + wave;            // 64 x 32 = 2048 tiles
  const int tm = tile >> 5, tn = tile & 31;
  const int m0 = tm * 64, n0 = tn * 64;
  const int l15 = lane & 15, hi = lane >> 4;
  const int KDIM = HH * DD;                          // 2048

  v8f acc[4][4];
#pragma unroll
  for (int i = 0; i < 4; ++i)
#pragma unroll
    for (int j = 0; j < 4; ++j) acc[i][j] = zero8();

  for (int k0 = 0; k0 < KDIM; k0 += 32) {
    __builtin_prefetch(ctxb + (size_t)(m0 + l15) * KDIM + k0 + 256, 0, 0);
    __builtin_prefetch(wob + (size_t)(n0 + l15) * KDIM + k0 + 256, 0, 0);
    v16bf af[4], bfr[4];
#pragma unroll
    for (int i = 0; i < 4; ++i) af[i] = load_frag(ctxb, m0 + 16 * i + l15, k0, KDIM, hi);
#pragma unroll
    for (int j = 0; j < 4; ++j) bfr[j] = load_frag(wob, n0 + 16 * j + l15, k0, KDIM, hi);
#pragma unroll
    for (int i = 0; i < 4; ++i)
#pragma unroll
      for (int j = 0; j < 4; ++j) acc[i][j] = wmma_bf16(af[i], bfr[j], acc[i][j]);
  }

#pragma unroll
  for (int i = 0; i < 4; ++i)
#pragma unroll
    for (int j = 0; j < 4; ++j)
#pragma unroll
      for (int r = 0; r < 8; ++r) {
        const int row = m0 + 16 * i + r + 8 * hi;
        out[(size_t)row * EE + n0 + 16 * j + l15] = acc[i][j][r];
      }
}

extern "C" void kernel_launch(void* const* d_in, const int* in_sizes, int n_in,
                              void* d_out, int out_size, void* d_ws, size_t ws_size,
                              hipStream_t stream) {
  const float* x    = (const float*)d_in[0];   // [2,2048,2048]
  const float* wqkv = (const float*)d_in[1];   // [3072,2048]
  const float* wo   = (const float*)d_in[2];   // [2048,2048]
  float* out = (float*)d_out;

  char* ws = (char*)d_ws;
  size_t off = 0;
  auto alloc = [&](size_t bytes) -> void* {
    void* p = ws + off;
    off += (bytes + 255) & ~(size_t)255;
    return p;
  };
  const size_t nX = (size_t)BS * EE;           // 8388608
  const size_t nW = (size_t)FQKV * EE;         // 6291456
  const size_t nWo = (size_t)EE * (HH * DD);   // 4194304
  const size_t nQ = (size_t)BB * HH * SS * DD; // 8388608
  const size_t nK = (size_t)BB * KVH * SS * DD;// 2097152
  uint16_t* xb    = (uint16_t*)alloc(nX * 2);
  uint16_t* wqkvb = (uint16_t*)alloc(nW * 2);
  uint16_t* wob   = (uint16_t*)alloc(nWo * 2);
  uint16_t* qb    = (uint16_t*)alloc(nQ * 2);
  uint16_t* kb    = (uint16_t*)alloc(nK * 2);
  uint16_t* vt    = (uint16_t*)alloc(nK * 2);
  uint16_t* ctxb  = (uint16_t*)alloc(nX * 2);

  cvt_bf16_kernel<<<(int)(nX / 256), 256, 0, stream>>>(x, xb, (int)nX);
  cvt_bf16_kernel<<<(int)(nW / 256), 256, 0, stream>>>(wqkv, wqkvb, (int)nW);
  cvt_bf16_kernel<<<(int)(nWo / 256), 256, 0, stream>>>(wo, wob, (int)nWo);

  qkv_gemm_kernel<<<768, 128, 0, stream>>>(xb, wqkvb, qb, kb, vt);   // 3072 tiles

  const int ropeThreads = (BB * (HH + KVH) * SS) * 64;               // 5242880
  rope_kernel<<<ropeThreads / 256, 256, 0, stream>>>(qb, kb);

  attn_kernel<<<512, 128, 0, stream>>>(qb, kb, vt, ctxb);            // 2048 waves

  out_gemm_kernel<<<512, 128, 0, stream>>>(ctxb, wob, out);          // 2048 tiles
}
